// TritonMoEMLP_31052613550637
// MI455X (gfx1250) — matrix operand order repaired and checked by reading, not compiled, not run on hardware
//
#include <hip/hip_runtime.h>
#include <hip/hip_bf16.h>

typedef _Float16 f16;
typedef __attribute__((ext_vector_type(16))) _Float16 v16h;
typedef __attribute__((ext_vector_type(8)))  _Float16 v8h;
typedef __attribute__((ext_vector_type(8)))  float    v8f;

#define T_TOK 2048
#define D_EMB 768
#define NEXP  64
#define EW    256
#define TOPK  8
#define TW    (NEXP * EW)   // 16384
#define MTILE 64            // tokens per grouped-GEMM block
#define MAXTILES 32         // ceil(2048/64): worst case all tokens on one expert
#define HSTRIDE 264         // 256 + 8 halves (16B pad) -> 2-way-minimal LDS banking

// ---------------------------------------------------------------------------
// fp32 -> f16 elementwise convert
// ---------------------------------------------------------------------------
__global__ void convert_f16_kernel(const float* __restrict__ in, f16* __restrict__ out, int n) {
    int i = blockIdx.x * blockDim.x + threadIdx.x;
    if (i < n) out[i] = (f16)in[i];
}

// ---------------------------------------------------------------------------
// Tiled transpose + convert: out[c*rows + r] = (f16) in[r*cols + c]
// ---------------------------------------------------------------------------
__global__ __launch_bounds__(256) void transpose_f16_kernel(
    const float* __restrict__ in, f16* __restrict__ out, int rows, int cols) {
    __shared__ float tile[32][33];
    int c0 = blockIdx.x * 32;
    int r0 = blockIdx.y * 32;
    int tx = threadIdx.x, ty = threadIdx.y;   // 32 x 8
    #pragma unroll
    for (int i = 0; i < 32; i += 8) {
        int r = r0 + ty + i, c = c0 + tx;
        tile[ty + i][tx] = (r < rows && c < cols) ? in[(size_t)r * cols + c] : 0.0f;
    }
    __syncthreads();
    #pragma unroll
    for (int i = 0; i < 32; i += 8) {
        int c = c0 + ty + i, r = r0 + tx;
        if (c < cols && r < rows) out[(size_t)c * rows + r] = (f16)tile[tx][ty + i];
    }
}

// ---------------------------------------------------------------------------
// Router: logits, sigmoid gates, top-8, bucket scatter, aux-loss accumulation.
// One block (64 threads = 2 wave32) per token.
// ---------------------------------------------------------------------------
__global__ __launch_bounds__(64) void moe_router_kernel(
    const float* __restrict__ x,      // [T, 768]
    const float* __restrict__ rw,     // [64, 768]
    int*   __restrict__ counts,       // [E]
    float* __restrict__ p_sum,        // [E]
    int*   __restrict__ btok,         // [E, T]
    float* __restrict__ bgate,        // [E, T]
    float* __restrict__ zloss_acc,    // scalar
    float* __restrict__ closs_acc)    // scalar
{
    __shared__ float logits[NEXP];
    __shared__ float probs[NEXP];
    const int t = blockIdx.x;
    const int e = threadIdx.x;

    const float* xr = x  + (size_t)t * D_EMB;
    const float* wr = rw + (size_t)e * D_EMB;
    float acc = 0.0f;
    #pragma unroll 4
    for (int d = 0; d < D_EMB; d += 4) {
        acc += xr[d] * wr[d] + xr[d+1] * wr[d+1] + xr[d+2] * wr[d+2] + xr[d+3] * wr[d+3];
    }
    logits[e] = acc;
    float p = 1.0f / (1.0f + __expf(-acc));
    probs[e] = p;
    atomicAdd(&p_sum[e], p);
    __syncthreads();

    if (e == 0) {
        // logsumexp^2 (z-loss) and sum-of-probs (compute loss)
        float mx = logits[0];
        for (int i = 1; i < NEXP; ++i) mx = fmaxf(mx, logits[i]);
        float se = 0.0f, sp = 0.0f;
        for (int i = 0; i < NEXP; ++i) { se += __expf(logits[i] - mx); sp += probs[i]; }
        float lse = mx + __logf(se);
        atomicAdd(zloss_acc, lse * lse * (1.0f / T_TOK));
        atomicAdd(closs_acc, sp * (1.0f / T_TOK));

        // top-8 (sigmoid is monotone -> top-k of probs == top-k of logits)
        int   sel_e[TOPK];
        float sel_w[TOPK];
        float wsum = 0.0f;
        for (int k = 0; k < TOPK; ++k) {
            int best = 0; float bv = probs[0];
            for (int i = 1; i < NEXP; ++i) { if (probs[i] > bv) { bv = probs[i]; best = i; } }
            sel_e[k] = best; sel_w[k] = bv;
            probs[best] = -1.0f;            // probs no longer needed; mutate in place
            wsum += bv;
        }
        wsum += 1e-20f;
        for (int k = 0; k < TOPK; ++k) {
            int ee  = sel_e[k];
            int pos = atomicAdd(&counts[ee], 1);
            btok [ee * T_TOK + pos] = t;
            bgate[ee * T_TOK + pos] = sel_w[k] / wsum;
        }
    }
}

// ---------------------------------------------------------------------------
// Grouped GEMM: one block per (expert, 64-token tile). 256 threads = 8 wave32.
// GEMM1: h = relu(x_tile @ w1_e)^2 * gate   (64 x 256), staged in LDS (f16)
//        per wave: 4 M-subtiles x 2 N-strips; B fragments shared across M.
// GEMM2: out += h @ w2_e                    (64 x 768), fp32 atomic combine
//        per wave: 6 N-strips, M-subtiles looped (B re-reads hit WGP cache).
// Weight layouts are n-major / K-contiguous f16 so each WMMA B fragment is a
// single contiguous 32-byte per-lane load; all f16 weights (50MB) stay in L2.
// ---------------------------------------------------------------------------
__global__ __launch_bounds__(256) void moe_grouped_gemm_kernel(
    const f16* __restrict__ xh,       // [T, 768]         row-major (K contiguous)
    const f16* __restrict__ w1t,      // [16384, 768]     n-major, K contiguous
    const f16* __restrict__ w2t,      // [768, 16384]     n-major, K contiguous
    const int*   __restrict__ counts, // [E]
    const int*   __restrict__ btok,   // [E, T]
    const float* __restrict__ bgate,  // [E, T]
    float* __restrict__ out)          // [T, 768]
{
    __shared__ f16   hshm[MTILE * HSTRIDE];   // ~33KB h tile (padded stride)
    __shared__ float sgate[MTILE];
    __shared__ int   stok[MTILE];

    const int e    = blockIdx.x >> 5;             // / MAXTILES
    const int tile = blockIdx.x & (MAXTILES - 1);
    const int cnt  = counts[e];
    if (tile * MTILE >= cnt) return;              // uniform exit: EXEC all-1s

    const int tid  = threadIdx.x;
    const int lane = tid & 31;
    const int wv   = tid >> 5;                    // 0..7
    const int hi   = lane >> 4;                   // lane half (K-group select)
    const int l15  = lane & 15;

    if (tid < MTILE) {
        int slot = tile * MTILE + tid;
        int ok   = slot < cnt;
        stok[tid]  = ok ? btok [e * T_TOK + slot] : -1;
        sgate[tid] = ok ? bgate[e * T_TOK + slot] : 0.0f;
    }
    __syncthreads();

    // ---------------- GEMM1: columns [wv*32, wv*32+32) of this expert --------
    {
        v8f acc[4][2];
        #pragma unroll
        for (int mt = 0; mt < 4; ++mt) { acc[mt][0] = (v8f){}; acc[mt][1] = (v8f){}; }

        const f16* arow[4];
        #pragma unroll
        for (int mt = 0; mt < 4; ++mt) {
            int tok = stok[mt * 16 + l15];
            if (tok < 0) tok = 0;                 // gate==0 zeroes padded rows
            arow[mt] = xh + (size_t)tok * D_EMB + hi * 16;
        }
        const f16* b0 = w1t + (size_t)(e * EW + wv * 32 + l15) * D_EMB + hi * 16;
        const f16* b1 = b0 + (size_t)16 * D_EMB;

        for (int kb = 0; kb < D_EMB; kb += 32) {
            // B fragments: column = l15, K = kb + hi*16 + [0,16) contiguous (32B)
            v16h bb0 = *(const v16h*)(b0 + kb);
            v16h bb1 = *(const v16h*)(b1 + kb);
            __builtin_prefetch(b0 + kb + 64, 0, 1);
            __builtin_prefetch(b1 + kb + 64, 0, 1);
            #pragma unroll
            for (int mt = 0; mt < 4; ++mt) {
                // A fragment: row = mt*16+l15, K = hi*16+[0,16) contiguous
                v8h alo = *(const v8h*)(arow[mt] + kb);
                v8h ahi = *(const v8h*)(arow[mt] + kb + 8);
                v16h a;
                #pragma unroll
                for (int i = 0; i < 8; ++i) { a[i] = alo[i]; a[i + 8] = ahi[i]; }
                acc[mt][0] = __builtin_amdgcn_wmma_f32_16x16x32_f16(false, a, false, bb0, (short)0, acc[mt][0], false, false);
                acc[mt][1] = __builtin_amdgcn_wmma_f32_16x16x32_f16(false, a, false, bb1, (short)0, acc[mt][1], false, false);
            }
        }
        // relu^2 * gate, write h tile to LDS as f16
        #pragma unroll
        for (int mt = 0; mt < 4; ++mt) {
            #pragma unroll
            for (int r = 0; r < 8; ++r) {
                int   m = mt * 16 + r + hi * 8;   // D-matrix row per §7.12.2
                float g = sgate[m];
                float v0 = acc[mt][0][r]; v0 = (v0 > 0.0f) ? v0 * v0 * g : 0.0f;
                float v1 = acc[mt][1][r]; v1 = (v1 > 0.0f) ? v1 * v1 * g : 0.0f;
                hshm[m * HSTRIDE + wv * 32 + l15]      = (f16)v0;
                hshm[m * HSTRIDE + wv * 32 + 16 + l15] = (f16)v1;
            }
        }
    }
    __syncthreads();

    // ---------------- GEMM2: 6 output strips of 16 per wave (8*96 = 768) ----
    const int nbase = wv * 96;
    const f16* bp[6];
    #pragma unroll
    for (int j = 0; j < 6; ++j) {
        bp[j] = w2t + (size_t)(nbase + j * 16 + l15) * TW + e * EW + hi * 16;
    }

    for (int mt = 0; mt < 4; ++mt) {
        v8f acc2[6];
        #pragma unroll
        for (int j = 0; j < 6; ++j) acc2[j] = (v8f){};

        const f16* ar = &hshm[(mt * 16 + l15) * HSTRIDE + hi * 16];
        for (int kb = 0; kb < EW; kb += 32) {
            v8h alo = *(const v8h*)(ar + kb);
            v8h ahi = *(const v8h*)(ar + kb + 8);
            v16h a;
            #pragma unroll
            for (int i = 0; i < 8; ++i) { a[i] = alo[i]; a[i + 8] = ahi[i]; }
            #pragma unroll
            for (int j = 0; j < 6; ++j) {
                v16h b = *(const v16h*)(bp[j] + kb);
                acc2[j] = __builtin_amdgcn_wmma_f32_16x16x32_f16(false, a, false, b, (short)0, acc2[j], false, false);
            }
        }
        #pragma unroll
        for (int r = 0; r < 8; ++r) {
            int m   = mt * 16 + r + hi * 8;
            int tok = stok[m];
            if (tok >= 0) {
                #pragma unroll
                for (int j = 0; j < 6; ++j) {
                    atomicAdd(&out[(size_t)tok * D_EMB + nbase + j * 16 + l15], acc2[j][r]);
                }
            }
        }
    }
}

// ---------------------------------------------------------------------------
// Finalize: f_i and load-balance loss (z-loss / compute-loss already atomically
// accumulated into d_out slots, which are zeroed at launch).
// ---------------------------------------------------------------------------
__global__ __launch_bounds__(64) void moe_finalize_kernel(
    const int* __restrict__ counts, const float* __restrict__ p_sum,
    float* __restrict__ out_scalars) // [0]=zloss [1]=lb [2]=closs [3..66]=f_i
{
    __shared__ float part[NEXP];
    int e = threadIdx.x;
    float fi = (float)counts[e] * (1.0f / (float)(T_TOK * TOPK));
    out_scalars[3 + e] = fi;
    part[e] = fi * (p_sum[e] * (1.0f / T_TOK));
    __syncthreads();
    if (e == 0) {
        float s = 0.0f;
        for (int i = 0; i < NEXP; ++i) s += part[i];
        out_scalars[1] = (float)NEXP * s;
    }
}

// ---------------------------------------------------------------------------
extern "C" void kernel_launch(void* const* d_in, const int* in_sizes, int n_in,
                              void* d_out, int out_size, void* d_ws, size_t ws_size,
                              hipStream_t stream) {
    const float* x  = (const float*)d_in[0];   // [1,2048,768]
    const float* rw = (const float*)d_in[1];   // [64,768]
    const float* w1 = (const float*)d_in[2];   // [768,16384]
    const float* w2 = (const float*)d_in[3];   // [16384,768]
    float* out = (float*)d_out;                // output, z, lb, closs, f_i

    // Workspace carve-up (256B aligned blocks): ~54.6 MB total
    char* ws = (char*)d_ws;
    size_t off = 0;
    auto carve = [&](size_t bytes) {
        void* p = ws + off;
        off = (off + bytes + 255) & ~(size_t)255;
        return p;
    };
    f16*   xh     = (f16*)  carve((size_t)T_TOK * D_EMB * 2);   // 3.0 MB
    f16*   w1t    = (f16*)  carve((size_t)TW * D_EMB * 2);      // 24 MB
    f16*   w2t    = (f16*)  carve((size_t)D_EMB * TW * 2);      // 24 MB
    int*   counts = (int*)  carve(NEXP * sizeof(int));
    float* p_sum  = (float*)carve(NEXP * sizeof(float));
    int*   btok   = (int*)  carve((size_t)NEXP * T_TOK * sizeof(int));
    float* bgate  = (float*)carve((size_t)NEXP * T_TOK * sizeof(float));
    (void)ws_size; (void)in_sizes; (void)n_in;

    float* scal = out + (size_t)T_TOK * D_EMB; // [z, lb, closs, f_i...]

    // Zero accumulation targets (graph-capture-safe async memsets)
    hipMemsetAsync(d_out, 0, (size_t)out_size * sizeof(float), stream);
    hipMemsetAsync(counts, 0, NEXP * sizeof(int), stream);
    hipMemsetAsync(p_sum,  0, NEXP * sizeof(float), stream);

    // Precision/layout conversion: all f16 weights (50MB) live in the 192MB L2
    {
        int n = T_TOK * D_EMB;
        convert_f16_kernel<<<(n + 255) / 256, 256, 0, stream>>>(x, xh, n);
    }
    transpose_f16_kernel<<<dim3(TW / 32, D_EMB / 32), dim3(32, 8), 0, stream>>>(w1, w1t, D_EMB, TW);
    transpose_f16_kernel<<<dim3(D_EMB / 32, TW / 32), dim3(32, 8), 0, stream>>>(w2, w2t, TW, D_EMB);

    // Router + bucket scatter + aux losses
    moe_router_kernel<<<T_TOK, 64, 0, stream>>>(x, rw, counts, p_sum, btok, bgate,
                                                &scal[0], &scal[2]);

    // Grouped expert GEMMs (gathered top-k, WMMA f16, 64-token tiles)
    moe_grouped_gemm_kernel<<<NEXP * MAXTILES, 256, 0, stream>>>(
        xh, w1t, w2t, counts, btok, bgate, out);

    // f_i + load-balance loss
    moe_finalize_kernel<<<1, 64, 0, stream>>>(counts, p_sum, scal);
}
